// NeRFMLP_43619687858544
// MI455X (gfx1250) — matrix operand order repaired and checked by reading, not compiled, not run on hardware
//
#include <hip/hip_runtime.h>
#include <cstdint>
#include <cstddef>

typedef _Float16 h16;
typedef __attribute__((ext_vector_type(16))) _Float16 v16h;
typedef __attribute__((ext_vector_type(8)))  float    v8f;
typedef __attribute__((ext_vector_type(4)))  unsigned u32x4;
typedef __attribute__((ext_vector_type(8)))  int      i32x8;

#define THREADS 128

// f16 weight blob offsets (element counts), layout per layer: [fan_out][K_pad]
static constexpr size_t OFF1 = 0;                    // 256 x 64
static constexpr size_t OFF2 = OFF1 + 256 * 64;      // 256 x 256
static constexpr size_t OFF3 = OFF2 + 256 * 256;
static constexpr size_t OFF4 = OFF3 + 256 * 256;
static constexpr size_t OFF5 = OFF4 + 256 * 256;     // 256 x 320
static constexpr size_t OFF6 = OFF5 + 256 * 320;
static constexpr size_t OFF7 = OFF6 + 256 * 256;
static constexpr size_t OFF8 = OFF7 + 256 * 256;
static constexpr size_t OFFF = OFF8 + 256 * 256;     // 256 x 256 (feat)
static constexpr size_t OFFC = OFFF + 256 * 256;     // 128 x 320 (color)

// ---------------------------------------------------------------------------
// Prep: convert fp32 [fan_in][fan_out] -> f16 transposed padded [fan_out][K_pad]
// ---------------------------------------------------------------------------
__global__ void prep_weights(const float* __restrict__ src, h16* __restrict__ dst,
                             int fan_in, int fan_out, int k_pad) {
  int idx = blockIdx.x * blockDim.x + threadIdx.x;
  if (idx >= fan_out * k_pad) return;
  int n = idx / k_pad;
  int k = idx - n * k_pad;
  float v = (k < fan_in) ? src[(size_t)k * fan_out + n] : 0.0f;
  dst[idx] = (h16)v;
}

// ---------------------------------------------------------------------------
// TDM: DMA a [rows][64] f16 tile (row stride = k_pad elements) from global
// into LDS at lds_off. 2-D descriptor per cdna5 CDNA5 ISA.
// Issued per-wave (EXEC ignored); completion tracked with TENSORcnt.
// ---------------------------------------------------------------------------
__device__ __forceinline__ void tdm_load_chunk(const void* gaddr, unsigned lds_off,
                                               int k_pad, int rows) {
  unsigned long long ga = (unsigned long long)(uintptr_t)gaddr;
  u32x4 g0;
  g0[0] = 1u;                                    // count=1 (valid descriptor)
  g0[1] = lds_off;                               // lds_addr        [63:32]
  g0[2] = (unsigned)ga;                          // global_addr     [95:64]
  g0[3] = ((unsigned)(ga >> 32) & 0x01FFFFFFu)   // global_addr     [120:96]
        | (2u << 30);                            // type=2 (image)
  i32x8 g1;
  g1[0] = (1 << 16);                             // data_size=1 (2B), wg_mask=0
  g1[1] = (k_pad << 16);                         // tensor_dim0 lo16 [63:48]
  g1[2] = (rows << 16);                          // tensor_dim0 hi=0, tensor_dim1 lo16
  g1[3] = (64 << 16);                            // tensor_dim1 hi=0, tile_dim0=64
  g1[4] = rows & 0xFFFF;                         // tile_dim1=rows, tile_dim2=0
  g1[5] = k_pad;                                 // tensor_dim0_stride lo32
  g1[6] = 0;                                     // stride hi, tensor_dim1_stride lo
  g1[7] = 0;
  asm volatile("tensor_load_to_lds %0, %1" :: "s"(g0), "s"(g1) : "memory");
}

// ---------------------------------------------------------------------------
// WMMA fragment loaders (wave32, 16x16x32 f16)
// A (16xK rows=rays): lanes 0-15 rows, kgroup split {kg..kg+7, kg+16..kg+23}
// B (Kx16 as W^T rows=out-features): same kgroup split as A
// ---------------------------------------------------------------------------
union Frag { v16h v; u32x4 q[2]; };

__device__ __forceinline__ v16h load_fragA(const h16* base, int stride, int row0, int k0) {
  int lane = (int)(threadIdx.x & 31u);
  const h16* p = base + (size_t)(row0 + (lane & 15)) * stride + k0 + ((lane >> 4) << 3);
  Frag f;
  f.q[0] = *(const u32x4*)(p);
  f.q[1] = *(const u32x4*)(p + 16);
  return f.v;
}

__device__ __forceinline__ v16h load_fragB(const h16* base, int stride, int col0, int k0) {
  int lane = (int)(threadIdx.x & 31u);
  const h16* p = base + (size_t)(col0 + (lane & 15)) * stride + k0 + ((lane >> 4) << 3);
  Frag f;
  f.q[0] = *(const u32x4*)(p);
  f.q[1] = *(const u32x4*)(p + 16);
  return f.v;
}

// ---------------------------------------------------------------------------
// One fused layer: out[64][NT*16] = act( in[64][K] @ W + b )
//   in0: main activation LDS buffer (stride0, covers k < k_main)
//   in1: tail LDS buffer (pos-enc / dir-enc skip input, k >= k_main)
//   weights: [NT*16][K_pad] f16 blob, TDM-streamed in 64-K chunks,
//            double-buffered in wbuf0/wbuf1; wave 0 drives the DMA.
// ---------------------------------------------------------------------------
template <int NT>
__device__ void mlp_layer(const h16* in0, int stride0, int k_main,
                          const h16* in1, int stride1,
                          const h16* __restrict__ wblob, int k_pad,
                          const float* __restrict__ bias, bool relu_act,
                          h16* out, int out_stride, h16* wbuf0, h16* wbuf1) {
  const int tid  = (int)threadIdx.x;
  const int lane = tid & 31;
  const int wave = tid >> 5;
  const int row0 = wave << 4;          // 16 rays per wave
  const bool issuer = (wave == 0);

  const v8f vzero = {0.f, 0.f, 0.f, 0.f, 0.f, 0.f, 0.f, 0.f};
  v8f acc[NT];
#pragma unroll
  for (int t = 0; t < NT; ++t) acc[t] = vzero;

  const char* gbase = (const char*)wblob;       // chunk c starts at +c*128 bytes
  const unsigned lds0 = (unsigned)(uintptr_t)wbuf0;
  const unsigned lds1 = (unsigned)(uintptr_t)wbuf1;
  const int nchunks = k_pad / 64;

  __syncthreads();                              // wbuf free (prev layer done)
  if (issuer) tdm_load_chunk(gbase, lds0, k_pad, NT * 16);

  for (int c = 0; c < nchunks; ++c) {
    const int k0 = c * 64;
    if (issuer) __builtin_amdgcn_s_wait_tensorcnt(0);
    __syncthreads();                            // chunk c resident in LDS
    if (issuer && (c + 1 < nchunks))            // prefetch next chunk via TDM
      tdm_load_chunk(gbase + (size_t)(c + 1) * 128,
                     ((c + 1) & 1) ? lds1 : lds0, k_pad, NT * 16);

    const h16* wb = (c & 1) ? wbuf1 : wbuf0;
    const h16* src; int sstride, kl;
    if (k0 < k_main) { src = in0; sstride = stride0; kl = k0; }
    else             { src = in1; sstride = stride1; kl = k0 - k_main; }

    // A fragments for both K-steps up front; B fragments rotated one tile
    // ahead so ds_load latency overlaps the previous WMMA (1 wave/SIMD here,
    // so intra-wave pipelining is the only latency cover).
    v16h a0 = load_fragA(src, sstride, row0, kl);
    v16h a1 = load_fragA(src, sstride, row0, kl + 32);
#pragma unroll
    for (int ks = 0; ks < 2; ++ks) {
      const v16h a = ks ? a1 : a0;
      v16h b = load_fragB(wb, 64, 0, ks * 32);
#pragma unroll
      for (int t = 0; t < NT; ++t) {
        const v16h cur = b;
        if (t + 1 < NT) b = load_fragB(wb, 64, (t + 1) * 16, ks * 32);
        acc[t] = __builtin_amdgcn_wmma_f32_16x16x32_f16(
            false, a, false, cur, (short)0, acc[t], false, false);
      }
    }
  }

  // epilogue: D layout -> lanes 0-15: n=lane, m=r ; lanes 16-31: n=lane-16, m=r+8
  const int m16 = lane & 15;
  const int hi8 = (lane >> 4) << 3;
#pragma unroll
  for (int t = 0; t < NT; ++t) {
    const int n = t * 16 + m16;
    const float bv = bias[n];
#pragma unroll
    for (int r = 0; r < 8; ++r) {
      float v = acc[t][r] + bv;
      if (relu_act) v = fmaxf(v, 0.0f);
      out[(size_t)(row0 + r + hi8) * out_stride + n] = (h16)v;
    }
  }
}

// ---------------------------------------------------------------------------
// Fully fused NeRF MLP: 64 rays per block, 4 waves, activations live in LDS,
// weights streamed by the Tensor Data Mover.
// ---------------------------------------------------------------------------
__global__ __launch_bounds__(THREADS, 1)
void nerf_fused(const float* __restrict__ pos, const float* __restrict__ dir,
                const h16* __restrict__ blob,
                const float* __restrict__ b1, const float* __restrict__ b2,
                const float* __restrict__ b3, const float* __restrict__ b4,
                const float* __restrict__ b5, const float* __restrict__ b6,
                const float* __restrict__ b7, const float* __restrict__ b8,
                const float* __restrict__ Ws, const float* __restrict__ bs,
                const float* __restrict__ bf, const float* __restrict__ bc,
                const float* __restrict__ Wr, const float* __restrict__ br,
                float* __restrict__ out, int N) {
  __shared__ __align__(16) h16 sA[64 * 256];
  __shared__ __align__(16) h16 sB[64 * 256];
  __shared__ __align__(16) h16 sPE[64 * 64];   // pos-enc, 63 used, padded
  __shared__ __align__(16) h16 sDE[64 * 64];   // dir-enc, 27 used, padded
  __shared__ __align__(16) h16 sW0[256 * 64];  // weight chunk, TDM buffer 0
  __shared__ __align__(16) h16 sW1[256 * 64];  // weight chunk, TDM buffer 1

  const int tid  = (int)threadIdx.x;
  const int ray0 = (int)blockIdx.x * 64;

  // --- positional / directional encoding (1 ray per thread) ---
  if (tid < 64) {
    int r = tid;
    float x = pos[(size_t)(ray0 + r) * 3 + 0];
    float y = pos[(size_t)(ray0 + r) * 3 + 1];
    float z = pos[(size_t)(ray0 + r) * 3 + 2];
    h16* pe = &sPE[r * 64];
    pe[0] = (h16)x; pe[1] = (h16)y; pe[2] = (h16)z;
    float f = 1.0f;
#pragma unroll
    for (int l = 0; l < 10; ++l) {
      pe[3 + 6 * l + 0] = (h16)__sinf(f * x);
      pe[3 + 6 * l + 1] = (h16)__sinf(f * y);
      pe[3 + 6 * l + 2] = (h16)__sinf(f * z);
      pe[3 + 6 * l + 3] = (h16)__cosf(f * x);
      pe[3 + 6 * l + 4] = (h16)__cosf(f * y);
      pe[3 + 6 * l + 5] = (h16)__cosf(f * z);
      f *= 2.0f;
    }
    pe[63] = (h16)0.0f;
  } else {
    int r = tid - 64;
    float x = dir[(size_t)(ray0 + r) * 3 + 0];
    float y = dir[(size_t)(ray0 + r) * 3 + 1];
    float z = dir[(size_t)(ray0 + r) * 3 + 2];
    h16* de = &sDE[r * 64];
    de[0] = (h16)x; de[1] = (h16)y; de[2] = (h16)z;
    float f = 1.0f;
#pragma unroll
    for (int l = 0; l < 4; ++l) {
      de[3 + 6 * l + 0] = (h16)__sinf(f * x);
      de[3 + 6 * l + 1] = (h16)__sinf(f * y);
      de[3 + 6 * l + 2] = (h16)__sinf(f * z);
      de[3 + 6 * l + 3] = (h16)__cosf(f * x);
      de[3 + 6 * l + 4] = (h16)__cosf(f * y);
      de[3 + 6 * l + 5] = (h16)__cosf(f * z);
      f *= 2.0f;
    }
    for (int i = 27; i < 64; ++i) de[i] = (h16)0.0f;
  }

  // --- MLP trunk ---
  mlp_layer<16>(sPE, 64, 64,  sPE, 64, blob + OFF1, 64,  b1, true,  sA, 256, sW0, sW1);
  mlp_layer<16>(sA, 256, 256, sA, 256, blob + OFF2, 256, b2, true,  sB, 256, sW0, sW1);
  mlp_layer<16>(sB, 256, 256, sB, 256, blob + OFF3, 256, b3, true,  sA, 256, sW0, sW1);
  mlp_layer<16>(sA, 256, 256, sA, 256, blob + OFF4, 256, b4, true,  sB, 256, sW0, sW1);
  mlp_layer<16>(sB, 256, 256, sPE, 64, blob + OFF5, 320, b5, true,  sA, 256, sW0, sW1); // skip: pos-enc
  mlp_layer<16>(sA, 256, 256, sA, 256, blob + OFF6, 256, b6, true,  sB, 256, sW0, sW1);
  mlp_layer<16>(sB, 256, 256, sB, 256, blob + OFF7, 256, b7, true,  sA, 256, sW0, sW1);
  mlp_layer<16>(sA, 256, 256, sA, 256, blob + OFF8, 256, b8, true,  sB, 256, sW0, sW1); // h -> sB
  __syncthreads();

  // --- sigma head: relu(h @ Ws + bs), out-dim 1 ---
  if (tid < 64) {
    float s = bs[0];
    const h16* h = &sB[tid * 256];
#pragma unroll 8
    for (int k = 0; k < 256; ++k) s += (float)h[k] * Ws[k];
    out[(size_t)3 * N + ray0 + tid] = fmaxf(s, 0.0f);
  }

  // --- feature + color layers ---
  mlp_layer<16>(sB, 256, 256, sB, 256, blob + OFFF, 256, bf, false, sA, 256, sW0, sW1); // feat
  mlp_layer<8> (sA, 256, 256, sDE, 64, blob + OFFC, 320, bc, true,  sB, 256, sW0, sW1); // skip: dir-enc
  __syncthreads();

  // --- rgb head: sigmoid(hc @ Wr + br), out-dim 3 ---
  if (tid < 64) {
    const h16* h = &sB[tid * 256];
#pragma unroll
    for (int c = 0; c < 3; ++c) {
      float s = br[c];
#pragma unroll 8
      for (int k = 0; k < 128; ++k) s += (float)h[k] * Wr[k * 3 + c];
      out[(size_t)(ray0 + tid) * 3 + c] = 1.0f / (1.0f + __expf(-s));
    }
  }
}

// ---------------------------------------------------------------------------
extern "C" void kernel_launch(void* const* d_in, const int* in_sizes, int n_in,
                              void* d_out, int out_size, void* d_ws, size_t ws_size,
                              hipStream_t stream) {
  (void)n_in; (void)out_size; (void)ws_size;
  const float* pos  = (const float*)d_in[0];
  const float* dirs = (const float*)d_in[1];
  h16* blob = (h16*)d_ws;

  struct P { int widx; int fi, fo, kp; size_t off; };
  const P L[10] = {
    { 2,  63, 256,  64, OFF1},
    { 4, 256, 256, 256, OFF2},
    { 6, 256, 256, 256, OFF3},
    { 8, 256, 256, 256, OFF4},
    {10, 319, 256, 320, OFF5},
    {12, 256, 256, 256, OFF6},
    {14, 256, 256, 256, OFF7},
    {16, 256, 256, 256, OFF8},
    {20, 256, 256, 256, OFFF},
    {22, 283, 128, 320, OFFC},
  };
  for (int i = 0; i < 10; ++i) {
    int tot = L[i].fo * L[i].kp;
    prep_weights<<<(tot + 255) / 256, 256, 0, stream>>>(
        (const float*)d_in[L[i].widx], blob + L[i].off, L[i].fi, L[i].fo, L[i].kp);
  }

  int N = in_sizes[0] / 3;
  nerf_fused<<<N / 64, THREADS, 0, stream>>>(
      pos, dirs, blob,
      (const float*)d_in[3],  (const float*)d_in[5],
      (const float*)d_in[7],  (const float*)d_in[9],
      (const float*)d_in[11], (const float*)d_in[13],
      (const float*)d_in[15], (const float*)d_in[17],
      (const float*)d_in[18], (const float*)d_in[19],
      (const float*)d_in[21], (const float*)d_in[23],
      (const float*)d_in[24], (const float*)d_in[25],
      (float*)d_out, N);
}
